// MultiHeadedAttention2_32504312496488
// MI455X (gfx1250) — matrix operand order repaired and checked by reading, not compiled, not run on hardware
//
#include <hip/hip_runtime.h>
#include <hip/hip_bf16.h>

typedef __attribute__((ext_vector_type(16))) _Float16 v16h;
typedef __attribute__((ext_vector_type(8)))  float    v8f;
typedef __attribute__((ext_vector_type(4)))  _Float16 h4;
typedef __attribute__((ext_vector_type(2)))  _Float16 h2;

#define NQ   1024
#define NK   1024
#define DH   256
#define KSTR 264   // Ksh row stride (halves): 528B -> bank step 4, conflict-free reads
#define VSTR 38    // Vsh row stride (halves): 76B -> bank step 19, conflict-free reads
#define PSTR 36
#define L2E  1.4426950408889634f

__global__ __launch_bounds__(256, 1)
void fattn_wmma_kernel(const float* __restrict__ q,
                       const float* __restrict__ kv,
                       float* __restrict__ out) {
    // double-buffered K/V staging (~81 KB total, WGP has 320 KB)
    __shared__ _Float16 Ksh[2][32 * KSTR];     // [buf][key][d]  (f16)
    __shared__ _Float16 Vsh[2][DH * VSTR];     // [buf][d][key]  (f16, transposed)
    __shared__ _Float16 Psh[8][16 * PSTR];     // per-wave P tile [m][k]

    const int tid  = threadIdx.x;
    const int lane = tid & 31;
    const int wave = tid >> 5;
    const int h    = lane >> 4;   // lane half
    const int ml   = lane & 15;

    const int batch = blockIdx.x >> 3;         // 8 q-blocks per batch
    const int qbase = (blockIdx.x & 7) << 7;   // *128
    const int qrow  = qbase + wave * 16;

    // ---- load Q fragments (A layout), pre-scaled by 1/sqrt(256) ----
    // per-lane A offsets are two contiguous runs of 8 floats: {8h+0..7}, {16+8h+0..7}
    const float* qp = q + ((size_t)batch * NQ + qrow + ml) * DH;
    const float sc = 0.0625f;
    v16h qf[8];
#pragma unroll
    for (int c = 0; c < 8; ++c) {
#pragma unroll
        for (int g = 0; g < 2; ++g) {
            const float4 t0 = *(const float4*)(qp + c * 32 + g * 16 + (h << 3));
            const float4 t1 = *(const float4*)(qp + c * 32 + g * 16 + (h << 3) + 4);
            qf[c][8 * g + 0] = (_Float16)(t0.x * sc);
            qf[c][8 * g + 1] = (_Float16)(t0.y * sc);
            qf[c][8 * g + 2] = (_Float16)(t0.z * sc);
            qf[c][8 * g + 3] = (_Float16)(t0.w * sc);
            qf[c][8 * g + 4] = (_Float16)(t1.x * sc);
            qf[c][8 * g + 5] = (_Float16)(t1.y * sc);
            qf[c][8 * g + 6] = (_Float16)(t1.z * sc);
            qf[c][8 * g + 7] = (_Float16)(t1.w * sc);
        }
    }

    v8f O[16];
#pragma unroll
    for (int n = 0; n < 16; ++n)
#pragma unroll
        for (int r = 0; r < 8; ++r) O[n][r] = 0.0f;

    float rm[8], rs[8];
#pragma unroll
    for (int r = 0; r < 8; ++r) { rm[r] = -1.0e30f; rs[r] = 0.0f; }

    const float* kvb = kv + (size_t)batch * NK * DH;

    for (int kc = 0; kc < NK / 32; ++kc) {
        const int p = kc & 1;
        // ---- stage 32 keys x 256 d: fp32 global -> f16 LDS, buf p ----
        // buf p was last read at chunk kc-2; chunk kc-1's barrier separates that from here.
        // Each thread handles 4 units; one unit = keys {2a,2a+1} x 4 d values.
        const float* kvp = kvb + (size_t)kc * 32 * DH;
        _Float16* Kp = Ksh[p];
        _Float16* Vp = Vsh[p];
#pragma unroll
        for (int u = 0; u < 4; ++u) {
            int i   = u * 256 + tid;     // 0..1023 = 16 key-pairs x 64 d-quads
            int a   = i >> 6;            // key pair
            int d4  = i & 63;            // d quad
            int key = a << 1;
            int d   = d4 << 2;
            const float4 t0 = *(const float4*)(kvp + (size_t)key * DH + d);
            const float4 t1 = *(const float4*)(kvp + (size_t)(key + 1) * DH + d);
            h4 k0, k1;
            k0.x = (_Float16)t0.x; k0.y = (_Float16)t0.y;
            k0.z = (_Float16)t0.z; k0.w = (_Float16)t0.w;
            k1.x = (_Float16)t1.x; k1.y = (_Float16)t1.y;
            k1.z = (_Float16)t1.z; k1.w = (_Float16)t1.w;
            *(h4*)&Kp[key * KSTR + d]       = k0;
            *(h4*)&Kp[(key + 1) * KSTR + d] = k1;
            h2 v0; v0.x = k0.x; v0.y = k1.x;
            h2 v1; v1.x = k0.y; v1.y = k1.y;
            h2 v2; v2.x = k0.z; v2.y = k1.z;
            h2 v3; v3.x = k0.w; v3.y = k1.w;
            *(h2*)&Vp[(d + 0) * VSTR + key] = v0;
            *(h2*)&Vp[(d + 1) * VSTR + key] = v1;
            *(h2*)&Vp[(d + 2) * VSTR + key] = v2;
            *(h2*)&Vp[(d + 3) * VSTR + key] = v3;
        }
        if (kc + 1 < NK / 32)
            __builtin_prefetch(kvb + (size_t)(kc + 1) * 32 * DH + tid * 32, 0, 0);
        __syncthreads();   // staging of buf p visible; prior chunk fully computed

        // ---- S = (Q*sc) @ K^T : two 16x16 tiles over K=256 in 8 steps ----
        v8f S[2];
#pragma unroll
        for (int t = 0; t < 2; ++t) {
            v8f acc = {0.f, 0.f, 0.f, 0.f, 0.f, 0.f, 0.f, 0.f};
#pragma unroll
            for (int c = 0; c < 8; ++c) {
                v16h bf;
#pragma unroll
                for (int v = 0; v < 8; ++v) {
                    int d = c * 32 + ((v & 4) ? 16 + ((v & 3) << 1) : (v << 1)) + (h << 3);      // B layout: table order
                    h2 pk = *(const h2*)&Kp[(t * 16 + ml) * KSTR + d];
                    bf[2 * v] = pk.x; bf[2 * v + 1] = pk.y;
                }
                acc = __builtin_amdgcn_wmma_f32_16x16x32_f16(
                    false, qf[c], false, bf, (short)0, acc, false, false);
            }
            S[t] = acc;
        }

        // ---- online softmax over the 32 new keys ----
        float corr[8];
#pragma unroll
        for (int r = 0; r < 8; ++r) {
            float mx = fmaxf(S[0][r], S[1][r]);
            mx = fmaxf(mx, __shfl_xor(mx, 1));
            mx = fmaxf(mx, __shfl_xor(mx, 2));
            mx = fmaxf(mx, __shfl_xor(mx, 4));
            mx = fmaxf(mx, __shfl_xor(mx, 8));
            float nm = fmaxf(rm[r], mx);
            float p0 = exp2f((S[0][r] - nm) * L2E);
            float p1 = exp2f((S[1][r] - nm) * L2E);
            S[0][r] = p0; S[1][r] = p1;
            float sum = p0 + p1;
            sum += __shfl_xor(sum, 1);
            sum += __shfl_xor(sum, 2);
            sum += __shfl_xor(sum, 4);
            sum += __shfl_xor(sum, 8);
            float cf = exp2f((rm[r] - nm) * L2E);
            rs[r] = rs[r] * cf + sum;
            rm[r] = nm;
            corr[r] = cf;
        }
#pragma unroll
        for (int n = 0; n < 16; ++n)
#pragma unroll
            for (int r = 0; r < 8; ++r) O[n][r] *= corr[r];

        // ---- P (C layout) -> per-wave LDS row-major [m][k] ----
#pragma unroll
        for (int t = 0; t < 2; ++t)
#pragma unroll
            for (int r = 0; r < 8; ++r)
                Psh[wave][(r + 8 * h) * PSTR + t * 16 + ml] = (_Float16)S[t][r];
        // wave-private buffer: same-wave LDS is in-order; fence compiler + DScnt
        asm volatile("s_wait_dscnt 0" ::: "memory");

        // ---- P fragment (A layout) ----
        v16h pf;
#pragma unroll
        for (int v = 0; v < 8; ++v) {
            int k = ((v & 4) ? 16 + ((v & 3) << 1) : (v << 1)) + (h << 3);
            h2 pk = *(const h2*)&Psh[wave][ml * PSTR + k];
            pf[2 * v] = pk.x; pf[2 * v + 1] = pk.y;
        }

        // ---- O += P @ V : 16 output column tiles ----
#pragma unroll
        for (int n = 0; n < 16; ++n) {
            v16h vf;
#pragma unroll
            for (int v = 0; v < 8; ++v) {
                int k = ((v & 4) ? 16 + ((v & 3) << 1) : (v << 1)) + (h << 3);                   // B layout: table order
                h2 pk = *(const h2*)&Vp[(n * 16 + ml) * VSTR + k];
                vf[2 * v] = pk.x; vf[2 * v + 1] = pk.y;
            }
            O[n] = __builtin_amdgcn_wmma_f32_16x16x32_f16(
                false, pf, false, vf, (short)0, O[n], false, false);
        }
    }

    // ---- normalize and store fp32 output ----
    float* op = out + ((size_t)batch * NQ + qrow) * DH;
#pragma unroll
    for (int r = 0; r < 8; ++r) {
        float inv = 1.0f / rs[r];
        int row = r + 8 * h;
#pragma unroll
        for (int n = 0; n < 16; ++n)
            op[row * DH + n * 16 + ml] = O[n][r] * inv;
    }
}

extern "C" void kernel_launch(void* const* d_in, const int* in_sizes, int n_in,
                              void* d_out, int out_size, void* d_ws, size_t ws_size,
                              hipStream_t stream) {
    const float* q  = (const float*)d_in[0];
    const float* kv = (const float*)d_in[1];
    float* out = (float*)d_out;
    // 64 batches x 8 q-blocks of 128 rows; 256 threads = 8 wave32s per block
    dim3 grid(64 * 8), block(256);
    fattn_wmma_kernel<<<grid, block, 0, stream>>>(q, kv, out);
}
